// SynapticPlasticityModule_33363305955749
// MI455X (gfx1250) — matrix operand order repaired and checked by reading, not compiled, not run on hardware
//
#include <hip/hip_runtime.h>
#include <hip/hip_bf16.h>

// ---------------------------------------------------------------------------
// SynapticPlasticityModule for MI455X (gfx1250, wave32, WMMA + async-LDS)
//
//   output   = x @ (W * meta)                      -- bf16x3 split WMMA GEMM
//   new_w    = clip(clip(W + outer(post[-1], a) - outer(b, pre[-1])) * s)
//   new_meta = clip(meta + mlr*(colmean(out) - meta), 0, 2)
//   hist     = activity_history with hist[ptr] = mean(out)
// ---------------------------------------------------------------------------

typedef __attribute__((ext_vector_type(16))) __bf16        v16bf;
typedef __attribute__((ext_vector_type(8)))  float         v8f;
typedef __attribute__((ext_vector_type(4)))  float         v4f;
typedef __attribute__((ext_vector_type(4)))  unsigned int  v4u;

#define HDIM   1024
#define HIST   100
#define AHLEN  1000

#define BM     128
#define BN     128
#define BK     32
#define LDSTR  40        // padded k-stride (bf16 elems): 80 B rows, 16B-aligned frags

// ---------------------------------------------------------------------------
// helpers
// ---------------------------------------------------------------------------
static __device__ __forceinline__ v16bf load_frag(const __bf16* p0, const __bf16* p1) {
    union { v4u q[2]; v16bf v; } u;
    u.q[0] = *reinterpret_cast<const v4u*>(p0);
    u.q[1] = *reinterpret_cast<const v4u*>(p1);
    return u.v;
}

// async 16-byte global -> LDS copy (ASYNCcnt-tracked, no VGPR data round-trip)
static __device__ __forceinline__ void async_cp16(unsigned lds_off, const void* g) {
    asm volatile("global_load_async_to_lds_b128 %0, %1, off"
                 :: "v"(lds_off), "v"((unsigned long long)g)
                 : "memory");
}
static __device__ __forceinline__ void wait_async0() {
    asm volatile("s_wait_asynccnt 0x0" ::: "memory");
}

// ---------------------------------------------------------------------------
// 0) zero scratch accumulators
// ---------------------------------------------------------------------------
__global__ void k_zero(float* __restrict__ p, int n) {
    int i = blockIdx.x * 256 + threadIdx.x;
    if (i < n) p[i] = 0.0f;
}

// ---------------------------------------------------------------------------
// 1) W' = W * meta (column scale), split to bf16 hi/lo AND transpose to
//    [n][k] so the GEMM's B tiles are row-contiguous (pure byte copies).
//    32x32 LDS-tiled transpose: coalesced reads AND coalesced writes.
// ---------------------------------------------------------------------------
__global__ __launch_bounds__(256)
void k_prep_wT(const float* __restrict__ w, const float* __restrict__ meta,
               __bf16* __restrict__ whiT, __bf16* __restrict__ wloT) {
    __shared__ __bf16 thi[32 * 34];
    __shared__ __bf16 tlo[32 * 34];
    const int tid = threadIdx.x;
    const int k0 = blockIdx.y * 32;
    const int n0 = blockIdx.x * 32;
#pragma unroll
    for (int e = 0; e < 4; ++e) {
        int idx = e * 256 + tid;            // 0..1023
        int kk = idx >> 5, nn = idx & 31;
        float v = w[(size_t)(k0 + kk) * HDIM + n0 + nn] * meta[n0 + nn];
        __bf16 hi = (__bf16)v;
        thi[kk * 34 + nn] = hi;
        tlo[kk * 34 + nn] = (__bf16)(v - (float)hi);
    }
    __syncthreads();
#pragma unroll
    for (int e = 0; e < 4; ++e) {
        int idx = e * 256 + tid;
        int nn = idx >> 5, kk = idx & 31;
        size_t g = (size_t)(n0 + nn) * HDIM + k0 + kk;
        whiT[g] = thi[kk * 34 + nn];
        wloT[g] = tlo[kk * 34 + nn];
    }
}

// ---------------------------------------------------------------------------
// 2) column-sum accumulator (used for colmean(x) and colmean(output))
// ---------------------------------------------------------------------------
__global__ __launch_bounds__(256)
void k_colsum(const float* __restrict__ m, float* __restrict__ sum, int rows_per_chunk) {
    int col = blockIdx.x * 256 + threadIdx.x;          // < HDIM
    int r0  = blockIdx.y * rows_per_chunk;
    float s = 0.0f;
    for (int r = 0; r < rows_per_chunk; ++r)
        s += m[(size_t)(r0 + r) * HDIM + col];
    atomicAdd(&sum[col], s);
}

// ---------------------------------------------------------------------------
// 3) bf16x3 split-precision WMMA GEMM:  out[B,H] = x[B,H] @ Wscaled[H,H]
//    block tile 128x128, BK=32, 8 wave32 waves, wave tile 32x64.
//    Double-buffered LDS; B tiles filled with async global->LDS B128 copies,
//    x tile converted fp32->bf16 hi/lo by VALU while WMMAs run on buffer p.
// ---------------------------------------------------------------------------
__global__ __launch_bounds__(256)
void k_gemm(const float* __restrict__ x,
            const __bf16* __restrict__ whiT, const __bf16* __restrict__ wloT,
            float* __restrict__ out) {
    __shared__ __bf16 Ahi[2][BM * LDSTR];
    __shared__ __bf16 Alo[2][BM * LDSTR];
    __shared__ __bf16 Bhi[2][BN * LDSTR];
    __shared__ __bf16 Blo[2][BN * LDSTR];

    const int tid  = threadIdx.x;
    const int lane = tid & 31;
    const int wave = tid >> 5;
    const int wm   = wave >> 1;      // 0..3 : 32-row M strip
    const int wn   = wave & 1;       // 0..1 : 64-col N strip
    const int m0   = blockIdx.y * BM;
    const int n0   = blockIdx.x * BN;

    const int lrow   = lane & 15;
    const int khalfA = (lane >> 4) * 8;    // A: lanes16-31 hold K=8..15 / 24..31
    const int khalfB = (lane >> 4) * 8;    // B: lanes16-31 hold K=8..15 / 24..31

    // stage one K-slab into buffer p
    auto stage = [&](int kt, int p) {
        // ---- x tile [BM x BK]: vectorized fp32 loads, bf16 hi/lo split ----
        {
            int m = tid >> 1, seg = tid & 1;          // 16 floats per thread
            const v4f* xs = (const v4f*)&x[(size_t)(m0 + m) * HDIM + kt + seg * 16];
            float fv[16];
            *(v4f*)&fv[0]  = xs[0];
            *(v4f*)&fv[4]  = xs[1];
            *(v4f*)&fv[8]  = xs[2];
            *(v4f*)&fv[12] = xs[3];
            union { __bf16 h[16]; v4u q[2]; } hi, lo;
#pragma unroll
            for (int e = 0; e < 16; ++e) {
                __bf16 h = (__bf16)fv[e];
                hi.h[e] = h;
                lo.h[e] = (__bf16)(fv[e] - (float)h);
            }
            v4u* dh = (v4u*)&Ahi[p][m * LDSTR + seg * 16];
            v4u* dl = (v4u*)&Alo[p][m * LDSTR + seg * 16];
            dh[0] = hi.q[0]; dh[1] = hi.q[1];
            dl[0] = lo.q[0]; dl[1] = lo.q[1];
        }
        // ---- W tiles [BN rows x 64B]: async global->LDS 16B copies ----
#pragma unroll
        for (int it = 0; it < 2; ++it) {
            int c = it * 256 + tid;                   // 0..511 chunks of 16B
            int n = c >> 2, q = c & 3;
            size_t g = (size_t)(n0 + n) * HDIM + kt + q * 8;
            async_cp16((unsigned)(size_t)&Bhi[p][n * LDSTR + q * 8], whiT + g);
            async_cp16((unsigned)(size_t)&Blo[p][n * LDSTR + q * 8], wloT + g);
        }
    };

    v8f c[2][4];
    const v8f vzero = {0.f, 0.f, 0.f, 0.f, 0.f, 0.f, 0.f, 0.f};
#pragma unroll
    for (int mt = 0; mt < 2; ++mt)
#pragma unroll
        for (int nt = 0; nt < 4; ++nt) c[mt][nt] = vzero;

    stage(0, 0);
    int p = 0;
    for (int kt = 0; kt < HDIM; kt += BK) {
        wait_async0();          // our async B-tile copies into buffer p done
        __syncthreads();        // everyone's stores to buffer p visible

        // fill the other buffer while this one is consumed
        if (kt + BK < HDIM) {
            stage(kt + BK, p ^ 1);
            // pull the x slab after next into L2 while we compute
            if (kt + 2 * BK < HDIM)
                __builtin_prefetch(&x[(size_t)(m0 + (tid >> 1)) * HDIM + kt + 2 * BK +
                                      (tid & 1) * 16], 0, 1);
        }

        // ---- fragments per CDNA5 16-bit A / B VGPR layouts ----
        v16bf ah[2], al[2], bh[4], bl[4];
#pragma unroll
        for (int mt = 0; mt < 2; ++mt) {
            int r = (wm * 32 + mt * 16 + lrow) * LDSTR;
            ah[mt] = load_frag(&Ahi[p][r + khalfA], &Ahi[p][r + 16 + khalfA]);
            al[mt] = load_frag(&Alo[p][r + khalfA], &Alo[p][r + 16 + khalfA]);
        }
#pragma unroll
        for (int nt = 0; nt < 4; ++nt) {
            int r = (wn * 64 + nt * 16 + lrow) * LDSTR;
            bh[nt] = load_frag(&Bhi[p][r + khalfB], &Bhi[p][r + khalfB + 16]);
            bl[nt] = load_frag(&Blo[p][r + khalfB], &Blo[p][r + khalfB + 16]);
        }
        // ---- 3-product compensated bf16 WMMA: hi*hi + hi*lo + lo*hi ----
#pragma unroll
        for (int mt = 0; mt < 2; ++mt)
#pragma unroll
            for (int nt = 0; nt < 4; ++nt) {
                c[mt][nt] = __builtin_amdgcn_wmma_f32_16x16x32_bf16(
                    false, ah[mt], false, bh[nt], (short)0, c[mt][nt], false, false);
                c[mt][nt] = __builtin_amdgcn_wmma_f32_16x16x32_bf16(
                    false, ah[mt], false, bl[nt], (short)0, c[mt][nt], false, false);
                c[mt][nt] = __builtin_amdgcn_wmma_f32_16x16x32_bf16(
                    false, al[mt], false, bh[nt], (short)0, c[mt][nt], false, false);
            }
        p ^= 1;
    }

    // ---- epilogue: C/D layout = VGPR r -> M=r (+8 for lanes 16-31), N=lane&15
    const int mbase = m0 + wm * 32 + (lane >> 4) * 8;
    const int nbase = n0 + wn * 64 + (lane & 15);
#pragma unroll
    for (int mt = 0; mt < 2; ++mt)
#pragma unroll
        for (int nt = 0; nt < 4; ++nt)
#pragma unroll
            for (int r = 0; r < 8; ++r)
                out[(size_t)(mbase + mt * 16 + r) * HDIM + nbase + nt * 16] = c[mt][nt][r];
}

// ---------------------------------------------------------------------------
// 4) small updates: means, STDP combination vectors, meta, hist, homeo scale
// ---------------------------------------------------------------------------
__global__ __launch_bounds__(256)
void k_small(const float* __restrict__ xsum, const float* __restrict__ osum,
             const float* __restrict__ pre_hist, const float* __restrict__ post_hist,
             const float* __restrict__ act_hist, const int* __restrict__ hptr,
             const float* __restrict__ meta,
             const float* __restrict__ p_win, const float* __restrict__ p_dec,
             const float* __restrict__ p_target, const float* __restrict__ p_hrate,
             const float* __restrict__ p_mlr,
             float* __restrict__ xmean, float* __restrict__ omean,
             float* __restrict__ avec, float* __restrict__ bvec,
             float* __restrict__ scal,
             float* __restrict__ out_meta, float* __restrict__ out_hist,
             int Brows) {
    __shared__ float red[256];
    const int tid = threadIdx.x;
    const float win = p_win[0], dec = p_dec[0];
    const float invB = 1.0f / (float)Brows;

    // column means + grand total of output
    float tsum = 0.0f;
    for (int j = tid; j < HDIM; j += 256) {
        omean[j] = osum[j] * invB;
        xmean[j] = xsum[j] * invB;
        tsum += osum[j];
    }
    red[tid] = tsum;
    __syncthreads();
    for (int s = 128; s > 0; s >>= 1) {
        if (tid < s) red[tid] += red[tid + s];
        __syncthreads();
    }
    const float total_mean = red[0] * invB / (float)HDIM;
    __syncthreads();

    // STDP: avec = pos_w @ pre,  bvec = neg_w @ post   (post[99] = omean)
    for (int j = tid; j < HDIM; j += 256) {
        float a = 0.0f, b = 0.0f;
#pragma unroll 4
        for (int i = 0; i <= HIST - 2; ++i) {
            float wi = 0.01f * __expf(-(float)(HIST - i) * win * dec);
            a += wi * pre_hist[(i + 1) * HDIM + j];
            if (i >= 1) b += wi * post_hist[(i + 1) * HDIM + j];
        }
        b += 0.01f * __expf(-(float)(1) * win * dec) * omean[j];   // i = HIST-1 term
        avec[j] = a;
        bvec[j] = b;
        float m  = meta[j];
        float nm = m + p_mlr[0] * (omean[j] - m);
        out_meta[j] = fminf(fmaxf(nm, 0.0f), 2.0f);
    }

    // hist output + homeostatic scale
    const int ptr = hptr[0];
    float hsum = 0.0f;
    for (int t = tid; t < AHLEN; t += 256) {
        float h = (t == ptr) ? total_mean : act_hist[t];
        out_hist[t] = h;
        hsum += h;
    }
    red[tid] = hsum;
    __syncthreads();
    for (int s = 128; s > 0; s >>= 1) {
        if (tid < s) red[tid] += red[tid + s];
        __syncthreads();
    }
    if (tid == 0) {
        float diff = p_target[0] - red[0] / (float)AHLEN;
        scal[0] = 1.0f + p_hrate[0] * diff;
    }
}

// ---------------------------------------------------------------------------
// 5) new_w = clip(clip(W + outer(omean, avec) - outer(bvec, xmean)) * s)
// ---------------------------------------------------------------------------
__global__ __launch_bounds__(256)
void k_neww(const float* __restrict__ w,
            const float* __restrict__ omean, const float* __restrict__ avec,
            const float* __restrict__ bvec,  const float* __restrict__ xmean,
            const float* __restrict__ scal,  float* __restrict__ outw) {
    int idx = blockIdx.x * 256 + threadIdx.x;           // H*H total
    int h = idx >> 10, j = idx & 1023;
    float v = w[idx] + omean[h] * avec[j] - bvec[h] * xmean[j];
    v = fminf(fmaxf(v, -1.0f), 1.0f);
    v *= scal[0];
    v = fminf(fmaxf(v, -1.0f), 1.0f);
    outw[idx] = v;
}

// ---------------------------------------------------------------------------
// launch
// ---------------------------------------------------------------------------
extern "C" void kernel_launch(void* const* d_in, const int* in_sizes, int n_in,
                              void* d_out, int out_size, void* d_ws, size_t ws_size,
                              hipStream_t stream) {
    const float* x      = (const float*)d_in[0];
    const float* w      = (const float*)d_in[1];
    const float* meta   = (const float*)d_in[2];
    const float* preh   = (const float*)d_in[3];
    const float* posth  = (const float*)d_in[4];
    const float* acth   = (const float*)d_in[5];
    const int*   hptr   = (const int*)  d_in[6];
    const float* win    = (const float*)d_in[7];
    const float* dec    = (const float*)d_in[8];
    const float* target = (const float*)d_in[9];
    const float* hrate  = (const float*)d_in[10];
    const float* mlr    = (const float*)d_in[11];

    const int Brows = in_sizes[0] / HDIM;               // 16384

    // workspace layout: [WhiT bf16 2MB][WloT bf16 2MB][float vectors]
    char* wsb = (char*)d_ws;
    __bf16* WhiT = (__bf16*)wsb;
    __bf16* WloT = WhiT + (size_t)HDIM * HDIM;
    float*  f    = (float*)(wsb + 2 * (size_t)HDIM * HDIM * sizeof(__bf16));
    float *xsum = f,            *osum = f + HDIM,
          *xmean = f + 2*HDIM,  *omean = f + 3*HDIM,
          *avec = f + 4*HDIM,   *bvec = f + 5*HDIM,
          *scal = f + 6*HDIM;

    // output layout (flat, reference return order)
    float* outp = (float*)d_out;
    float* out_out  = outp;
    float* out_w    = outp + (size_t)Brows * HDIM;
    float* out_meta = out_w + (size_t)HDIM * HDIM;
    float* out_hist = out_meta + HDIM;

    k_zero   <<<(2 * HDIM + 255) / 256, 256, 0, stream>>>(xsum, 2 * HDIM);
    k_prep_wT<<<dim3(HDIM / 32, HDIM / 32), 256, 0, stream>>>(w, meta, WhiT, WloT);
    k_colsum <<<dim3(HDIM / 256, Brows / 128), 256, 0, stream>>>(x, xsum, 128);
    k_gemm   <<<dim3(HDIM / BN, Brows / BM), 256, 0, stream>>>(x, WhiT, WloT, out_out);
    k_colsum <<<dim3(HDIM / 256, Brows / 128), 256, 0, stream>>>(out_out, osum, 128);
    k_small  <<<1, 256, 0, stream>>>(xsum, osum, preh, posth, acth, hptr, meta,
                                     win, dec, target, hrate, mlr,
                                     xmean, omean, avec, bvec, scal,
                                     out_meta, out_hist, Brows);
    k_neww   <<<(HDIM * HDIM) / 256, 256, 0, stream>>>(w, omean, avec, bvec, xmean,
                                                       scal, out_w);
}
